// TrajectoryCrossAttention_27281632264946
// MI455X (gfx1250) — matrix operand order repaired and checked by reading, not compiled, not run on hardware
//
#include <hip/hip_runtime.h>

// ---------------------------------------------------------------------------
// TrajectoryCrossAttention for MI455X (gfx1250, wave32, WMMA).
// All GEMM-shaped work uses v_wmma_f32_16x16x32_f16 (f16 in, f32 accum),
// register-blocked 2x2 (32x32 output per wave).  __launch_bounds__(256,1)
// gives the WMMA kernels the full VGPR budget: round-2 codegen showed the
// 2x2 accumulators spilling to scratch inside the k-loop under the default
// occupancy-driven register cap, which defeats the blocking.
// Key algebraic simplification: reference einsum 'bhqf,bhsfd->bhqd' sums v2
// over its query axis s, so q_out = attn2 @ V2sum with
// V2sum = (sum_q x) @ Wpkv_v  -> the 59-GF v2 GEMM collapses to ~19M MACs.
// ---------------------------------------------------------------------------

typedef __attribute__((ext_vector_type(16))) _Float16 v16h;
typedef __attribute__((ext_vector_type(8)))  float    v8f;

#define BQ 2
#define QN 1568
#define CN 768
#define HN 12
#define DN 64
#define FNUM 8
#define SN 196
#define NN (FNUM * SN)   // 1568
#define SCALE 0.125f     // d^-0.5, d=64

static __device__ __forceinline__ v8f wmma16(v16h a, v16h b, v8f c) {
  // 8 args: (neg_a, A, neg_b, B, c_mod, C, reuse_a, reuse_b)
  return __builtin_amdgcn_wmma_f32_16x16x32_f16(false, a, false, b, (short)0, c,
                                                false, false);
}

// Load a 16-half A fragment (contiguous k order) from a row pointer:
// lane&15 selects the row (via the pointer), kA = (lane<16)?0:16,
// per-lane halves are the row's k = {kA..kA+15}.
static __device__ __forceinline__ v16h loadA(const float* __restrict__ ap) {
  v16h a;
#pragma unroll
  for (int j = 0; j < 8; ++j) { a[j] = (_Float16)ap[j]; a[8 + j] = (_Float16)ap[8 + j]; }
  return a;
}

// ---------------------------------------------------------------------------
// C[M,N] = scale * (A[M,K] @ B[K,N]) + bias.  One wave per 32x32 tile (2x2
// WMMA register blocking).  M,N multiples of 32; K multiple of 32 (K=768).
// B fragment per lane: col = lane&15 (+0/+16), 16 contiguous k starting at
// (lane<16)?0:16.  C fragment: acc[i] -> row base + i + 8*(lane>=16).
// ---------------------------------------------------------------------------
__global__ void __launch_bounds__(256, 1)
gemm_nn_f16(const float* __restrict__ A, const float* __restrict__ B,
            const float* __restrict__ bias, float* __restrict__ C,
            int tilesM, int tilesN, int K,
            int lda, int ldb, int ldc, float scaleOut)
{
  int wave = threadIdx.x >> 5;
  int lane = threadIdx.x & 31;
  int flat = blockIdx.x * 8 + wave;
  if (flat >= tilesM * tilesN) return;          // wave-uniform: EXEC stays full
  int tm = flat / tilesN, tn = flat % tilesN;
  int l15 = lane & 15;
  int hi  = (lane >> 4) & 1;
  int kA = hi * 16, kB = hi * 16;
  int col0 = tn * 32 + l15;
  const float* Arow0 = A + (size_t)(tm * 32 + l15) * lda;
  const float* Arow1 = Arow0 + (size_t)16 * lda;
  v8f acc00 = {}, acc01 = {}, acc10 = {}, acc11 = {};
  for (int k0 = 0; k0 < K; k0 += 32) {
    // gfx1250 prefetch of next k-panel (speculative; safe past end)
    __builtin_prefetch((const void*)(Arow0 + k0 + 32 + kA), 0, 1);
    __builtin_prefetch((const void*)(B + (size_t)(k0 + 32 + kB) * ldb + col0), 0, 1);
    v16h a0 = loadA(Arow0 + k0 + kA);
    v16h a1 = loadA(Arow1 + k0 + kA);
    const float* bp = B + (size_t)(k0 + kB) * ldb + col0;
    v16h b0, b1;
#pragma unroll
    for (int j = 0; j < 16; ++j) {
      b0[j] = (_Float16)bp[0];
      b1[j] = (_Float16)bp[16];
      bp += ldb;
    }
    acc00 = wmma16(a0, b0, acc00);
    acc01 = wmma16(a0, b1, acc01);
    acc10 = wmma16(a1, b0, acc10);
    acc11 = wmma16(a1, b1, acc11);
  }
  float bv0 = bias ? bias[col0] : 0.0f;
  float bv1 = bias ? bias[col0 + 16] : 0.0f;
  int r0 = tm * 32 + hi * 8;
#pragma unroll
  for (int i = 0; i < 8; ++i) {
    float* c0 = C + (size_t)(r0 + i) * ldc + col0;
    float* c1 = C + (size_t)(r0 + 16 + i) * ldc + col0;
    c0[0]  = acc00[i] * scaleOut + bv0;
    c0[16] = acc01[i] * scaleOut + bv1;
    c1[0]  = acc10[i] * scaleOut + bv0;
    c1[16] = acc11[i] * scaleOut + bv1;
  }
}

// ---------------------------------------------------------------------------
// Stage-1 scores: scores[bh][q][n] = 0.125 * sum_d q[b,q,h*64+d]*k[b,n,h*64+d]
// A @ B^T with K=64 (2 k-steps), 2x2 register blocking (32x32 per wave).
// Per-lane B loads are contiguous rows of k.  Written straight to output #2
// (pre-softmax scaled scores) -> HBM-store bound, ~236 MB.
// ---------------------------------------------------------------------------
__global__ void __launch_bounds__(256, 1)
scores_nt_f16(const float* __restrict__ q, const float* __restrict__ kv,
              float* __restrict__ scores)
{
  int wave = threadIdx.x >> 5, lane = threadIdx.x & 31;
  int bh = blockIdx.z;
  int b = bh / HN, h = bh % HN;
  const int tiles = (QN / 32) * (NN / 32);      // 49*49
  int flat = blockIdx.x * 8 + wave;
  if (flat >= tiles) return;                    // wave-uniform
  int tm = flat / (NN / 32), tn = flat % (NN / 32);
  int l15 = lane & 15, hi = (lane >> 4) & 1;
  int kA = hi * 16, kB = hi * 16;
  const float* Arow0 = q  + ((size_t)b * QN + tm * 32 + l15) * CN + h * DN;
  const float* Arow1 = Arow0 + (size_t)16 * CN;
  const float* Brow0 = kv + ((size_t)b * NN + tn * 32 + l15) * (2 * CN) + h * DN;
  const float* Brow1 = Brow0 + (size_t)16 * (2 * CN);
  v8f acc00 = {}, acc01 = {}, acc10 = {}, acc11 = {};
#pragma unroll
  for (int k0 = 0; k0 < DN; k0 += 32) {
    v16h a0 = loadA(Arow0 + k0 + kA);
    v16h a1 = loadA(Arow1 + k0 + kA);
    v16h b0, b1;
    const float* bp0 = Brow0 + k0 + kB;
    const float* bp1 = Brow1 + k0 + kB;
#pragma unroll
    for (int j = 0; j < 16; ++j) { b0[j] = (_Float16)bp0[j]; b1[j] = (_Float16)bp1[j]; }
    acc00 = wmma16(a0, b0, acc00);
    acc01 = wmma16(a0, b1, acc01);
    acc10 = wmma16(a1, b0, acc10);
    acc11 = wmma16(a1, b1, acc11);
  }
  float* out = scores + (size_t)bh * QN * NN;
  int r0 = tm * 32 + hi * 8, col0 = tn * 32 + l15;
#pragma unroll
  for (int i = 0; i < 8; ++i) {
    float* o0 = out + (size_t)(r0 + i) * NN + col0;
    float* o1 = out + (size_t)(r0 + 16 + i) * NN + col0;
    o0[0]  = SCALE * acc00[i];
    o0[16] = SCALE * acc01[i];
    o1[0]  = SCALE * acc10[i];
    o1[16] = SCALE * acc11[i];
  }
}

// ---------------------------------------------------------------------------
// Fused softmax(S=196) + P@V per (b,h,f): block handles 32 q-rows.
// Probabilities are built in LDS as f16 (zero-padded to K=224), then 8 waves
// (2 q-subtiles x 4 d-tiles) run the WMMA P@V with predicated V loads.
// x[b,q,f,h*64+dd] output.
// ---------------------------------------------------------------------------
#define PSTR 226   // odd dword count -> conflict-free row stride
__global__ void __launch_bounds__(256, 1)
softmax_pv_f16(const float* __restrict__ scores,
               const float* __restrict__ kv,
               float* __restrict__ x)
{
  __shared__ _Float16 probs[32][PSTR];
  int wave = threadIdx.x >> 5, lane = threadIdx.x & 31;
  int bh = blockIdx.z; int b = bh / HN, h = bh % HN;
  int f = blockIdx.y;
  int q0 = blockIdx.x * 32;
  const float* srow0 = scores + (size_t)bh * QN * NN + (size_t)q0 * NN + f * SN;

  // softmax: each wave owns 4 rows
  for (int rr = 0; rr < 4; ++rr) {
    int r = wave * 4 + rr;
    const float* sr = srow0 + (size_t)r * NN;
    float v[7], m = -3.0e38f;
#pragma unroll
    for (int t = 0; t < 7; ++t) {
      int s = lane + 32 * t;
      v[t] = (s < SN) ? sr[s] : -3.0e38f;
      m = fmaxf(m, v[t]);
    }
    for (int off = 16; off > 0; off >>= 1) m = fmaxf(m, __shfl_xor(m, off));
    float e[7], sum = 0.0f;
#pragma unroll
    for (int t = 0; t < 7; ++t) {
      int s = lane + 32 * t;
      e[t] = (s < SN) ? __expf(v[t] - m) : 0.0f;
      sum += e[t];
    }
    for (int off = 16; off > 0; off >>= 1) sum += __shfl_xor(sum, off);
    float inv = 1.0f / sum;
#pragma unroll
    for (int t = 0; t < 7; ++t) probs[r][lane + 32 * t] = (_Float16)(e[t] * inv);
  }
  __syncthreads();

  // P @ V : wave -> (qsub, d-tile)
  int qsub = wave >> 2, nt = wave & 3;
  int l15 = lane & 15, hi = (lane >> 4) & 1;
  int lr = qsub * 16 + l15;
  int dd = nt * 16 + l15;
  int kA = hi * 16, kB = hi * 16;
  const float* vbase = kv + ((size_t)b * NN + (size_t)f * SN) * (2 * CN)
                          + CN + h * DN + dd;
  v8f acc = {};
#pragma unroll
  for (int k0 = 0; k0 < 224; k0 += 32) {
    v16h a, bf;
#pragma unroll
    for (int j = 0; j < 8; ++j) {
      a[j]     = probs[lr][k0 + kA + j];
      a[8 + j] = probs[lr][k0 + kA + 8 + j];
    }
#pragma unroll
    for (int j = 0; j < 16; ++j) {
      int s = k0 + kB + j;
      bf[j] = (s < SN) ? (_Float16)vbase[(size_t)s * (2 * CN)] : (_Float16)0.0f;
    }
    acc = wmma16(a, bf, acc);
  }
  int rbase = q0 + qsub * 16 + hi * 8;
  int c = h * DN + dd;
#pragma unroll
  for (int i = 0; i < 8; ++i) {
    int qq = rbase + i;
    x[(((size_t)b * QN + qq) * FNUM + f) * CN + c] = acc[i];
  }
}

// Xsum[b,f,c] = sum_q x[b,q,f,c]
__global__ void xsum_kernel(const float* __restrict__ x, float* __restrict__ xsum)
{
  int t = blockIdx.x * 256 + threadIdx.x;
  if (t >= BQ * FNUM * CN) return;
  int c = t % CN, f = (t / CN) % FNUM, b = t / (CN * FNUM);
  const float* p = x + (((size_t)b * QN) * FNUM + f) * CN + c;
  float s = 0.0f;
  for (int qq = 0; qq < QN; ++qq) s += p[(size_t)qq * FNUM * CN];
  xsum[t] = s;
}

// V2sum[b,f,c'] = Xsum[b,f,:] @ Wpkv[:, C+c']   (tiny, fp32 VALU)
__global__ void v2sum_kernel(const float* __restrict__ xsum,
                             const float* __restrict__ Wpkv,
                             float* __restrict__ v2s)
{
  int t = blockIdx.x * 256 + threadIdx.x;
  if (t >= BQ * FNUM * CN) return;
  int cp = t % CN, bf = t / CN;
  const float* xr = xsum + (size_t)bf * CN;
  float s = 0.0f;
  for (int c = 0; c < CN; ++c) s += xr[c] * Wpkv[(size_t)c * (2 * CN) + CN + cp];
  v2s[t] = s;
}

// attn2 softmax over F=8 frames + y = attn2 @ V2sum   (per (b,q,h) thread)
__global__ void attn2_y_kernel(const float* __restrict__ q2, const float* __restrict__ k2,
                               const float* __restrict__ v2s, float* __restrict__ y)
{
  int t = blockIdx.x * 256 + threadIdx.x;
  if (t >= BQ * QN * HN) return;
  int h = t % HN, qq = (t / HN) % QN, b = t / (HN * QN);
  const float* q2r = q2 + ((size_t)b * QN + qq) * CN + h * DN;   // already *scale
  const float* k2b = k2 + (((size_t)b * QN + qq) * FNUM) * CN + h * DN;
  float logit[FNUM], m = -3.0e38f;
  for (int f = 0; f < FNUM; ++f) {
    const float* kr = k2b + (size_t)f * CN;
    float s = 0.0f;
    for (int d = 0; d < DN; ++d) s += q2r[d] * kr[d];
    logit[f] = s; m = fmaxf(m, s);
  }
  float sum = 0.0f;
  for (int f = 0; f < FNUM; ++f) { logit[f] = __expf(logit[f] - m); sum += logit[f]; }
  float inv = 1.0f / sum;
  float* yr = y + ((size_t)b * QN + qq) * CN + h * DN;
  for (int d = 0; d < DN; ++d) {
    float acc = 0.0f;
    for (int f = 0; f < FNUM; ++f)
      acc += logit[f] * inv * v2s[((size_t)b * FNUM + f) * CN + h * DN + d];
    yr[d] = acc;
  }
}

extern "C" void kernel_launch(void* const* d_in, const int* in_sizes, int n_in,
                              void* d_out, int out_size, void* d_ws, size_t ws_size,
                              hipStream_t stream) {
  (void)in_sizes; (void)n_in; (void)out_size; (void)ws_size;
  const float* query  = (const float*)d_in[0];
  const float* memory = (const float*)d_in[1];
  const float* Wq     = (const float*)d_in[2];
  const float* Wkv    = (const float*)d_in[3];
  const float* Wpq    = (const float*)d_in[4];
  const float* Wpkv   = (const float*)d_in[5];
  const float* Wproj  = (const float*)d_in[6];
  const float* bproj  = (const float*)d_in[7];

  float* out       = (float*)d_out;                      // q_out [B,Q,C]
  float* outScores = out + (size_t)BQ * QN * CN;         // scores [B*h,Q,F,S]

  float* ws  = (float*)d_ws;                             // ~202 MB total
  float* q   = ws;                                       // [B*Q, C]
  float* kvb = q   + (size_t)BQ * QN * CN;               // [B*N, 2C]
  float* q2  = kvb + (size_t)BQ * NN * 2 * CN;           // [B*Q, C] (pre-scaled)
  float* x   = q2  + (size_t)BQ * QN * CN;               // [B,Q,F,C]
  float* k2  = x   + (size_t)BQ * QN * FNUM * CN;        // [B*Q*F, C]
  float* xs  = k2  + (size_t)BQ * QN * FNUM * CN;        // [B,F,C]
  float* v2s = xs  + (size_t)BQ * FNUM * CN;             // [B,F,C]
  float* y   = v2s + (size_t)BQ * FNUM * CN;             // [B*Q, C]

  // q = query @ Wq
  { int tm = (BQ * QN) / 32, tn = CN / 32, tiles = tm * tn;
    gemm_nn_f16<<<(tiles + 7) / 8, 256, 0, stream>>>(query, Wq, nullptr, q,
        tm, tn, CN, CN, CN, CN, 1.0f); }
  // kv = memory @ Wkv
  { int tm = (BQ * NN) / 32, tn = (2 * CN) / 32, tiles = tm * tn;
    gemm_nn_f16<<<(tiles + 7) / 8, 256, 0, stream>>>(memory, Wkv, nullptr, kvb,
        tm, tn, CN, CN, 2 * CN, 2 * CN, 1.0f); }
  // q2 = scale * (q @ Wpq)
  { int tm = (BQ * QN) / 32, tn = CN / 32, tiles = tm * tn;
    gemm_nn_f16<<<(tiles + 7) / 8, 256, 0, stream>>>(q, Wpq, nullptr, q2,
        tm, tn, CN, CN, CN, CN, SCALE); }
  // pre-softmax scaled scores -> output #2
  { dim3 g(((QN / 32) * (NN / 32) + 7) / 8, 1, BQ * HN);
    scores_nt_f16<<<g, 256, 0, stream>>>(q, kvb, outScores); }
  // softmax over S + P@V -> x
  { dim3 g(QN / 32, FNUM, BQ * HN);
    softmax_pv_f16<<<g, 256, 0, stream>>>(outScores, kvb, x); }
  // k2 = x @ Wpkv[:, :C]
  { int tm = (BQ * QN * FNUM) / 32, tn = CN / 32, tiles = tm * tn;
    gemm_nn_f16<<<(tiles + 7) / 8, 256, 0, stream>>>(x, Wpkv, nullptr, k2,
        tm, tn, CN, CN, 2 * CN, CN, 1.0f); }
  // V2sum via Xsum trick
  xsum_kernel <<<(BQ * FNUM * CN + 255) / 256, 256, 0, stream>>>(x, xs);
  v2sum_kernel<<<(BQ * FNUM * CN + 255) / 256, 256, 0, stream>>>(xs, Wpkv, v2s);
  // temporal attention + y
  attn2_y_kernel<<<(BQ * QN * HN + 255) / 256, 256, 0, stream>>>(q2, k2, v2s, y);
  // out = y @ Wproj + bproj
  { int tm = (BQ * QN) / 32, tn = CN / 32, tiles = tm * tn;
    gemm_nn_f16<<<(tiles + 7) / 8, 256, 0, stream>>>(y, Wproj, bproj, out,
        tm, tn, CN, CN, CN, CN, 1.0f); }
}
